// DSB_7352984011529
// MI455X (gfx1250) — compile-verified, matrix operand order checked by reading, not run on hardware
//
#include <hip/hip_runtime.h>
#include <math.h>

// ---------- CDNA5 WMMA types ----------
typedef __attribute__((ext_vector_type(16))) __bf16        v16bf;
typedef __attribute__((ext_vector_type(8)))  float         v8f;
typedef __attribute__((ext_vector_type(8)))  unsigned int  v8u;
typedef __attribute__((ext_vector_type(4)))  unsigned int  u32x4;
typedef __attribute__((ext_vector_type(8)))  int           i32x8;
typedef __attribute__((ext_vector_type(4)))  int           i32x4;

// Problem constants (B=8, C=256, H=W=64)
#define HW       4096
#define NPIX     32768           // B*H*W
#define PER_B    1048576         // C*H*W
#define SLAB_STR 264             // LDS row stride (bf16) for a 64x256 A slab; 528B rows, 16B aligned

__device__ __forceinline__ float bf2f(unsigned short h) {
  unsigned int u = ((unsigned int)h) << 16;
  return __builtin_bit_cast(float, u);
}
__device__ __forceinline__ unsigned short f2bf(float f) {
  unsigned int u = __builtin_bit_cast(unsigned int, f);
  u += 0x7fffu + ((u >> 16) & 1u);              // round-to-nearest-even
  return (unsigned short)(u >> 16);
}

// Build a 16x32 (A) bf16 WMMA fragment from LDS slab row, K-step ki.
// ISA 7.12.2: lanes 0-15 -> K dword-pairs {0..7,16..23}, lanes 16-31 -> {8..15,24..31}.
__device__ __forceinline__ v16bf frag_ld(const unsigned short* lds, int row, int g, int ki) {
  union { v8u u; v16bf b; } f;
  const unsigned int* p32 = (const unsigned int*)(lds + row * SLAB_STR + ki * 32);
  int base = g * 4;
#pragma unroll
  for (int i = 0; i < 4; ++i) {
    f.u[i]     = p32[base + i];       // K = g*8 + 2i
    f.u[4 + i] = p32[base + 8 + i];   // K = 16 + g*8 + 2i
  }
  return f.b;
}

// ---------- per-batch mean / rstd over C*H*W ----------
__global__ void stats_kernel(const float* __restrict__ x, float* __restrict__ stats) {
  __shared__ float ssum[256], ssq[256];
  const int b = blockIdx.x;
  const float4* p = (const float4*)(x + (size_t)b * PER_B);
  float s = 0.f, q = 0.f;
  for (int i = threadIdx.x; i < PER_B / 4; i += 256) {
    float4 v = p[i];
    s += v.x + v.y + v.z + v.w;
    q += v.x * v.x + v.y * v.y + v.z * v.z + v.w * v.w;
  }
  ssum[threadIdx.x] = s; ssq[threadIdx.x] = q;
  __syncthreads();
  for (int off = 128; off > 0; off >>= 1) {
    if (threadIdx.x < off) { ssum[threadIdx.x] += ssum[threadIdx.x + off]; ssq[threadIdx.x] += ssq[threadIdx.x + off]; }
    __syncthreads();
  }
  if (threadIdx.x == 0) {
    float mu  = ssum[0] / (float)PER_B;
    float var = ssq[0] / (float)PER_B - mu * mu;
    stats[2 * b]     = mu;
    stats[2 * b + 1] = rsqrtf(var + 1e-5f);
  }
}

// ---------- normalize NCHW f32 -> NHWC bf16 via 64x64 LDS tile transpose ----------
__global__ void norm_pack(const float* __restrict__ x, const float* __restrict__ stats,
                          const float* __restrict__ gw, const float* __restrict__ gb,
                          unsigned short* __restrict__ out) {
  __shared__ unsigned short tile[64 * 66];         // pad 66: 33-dword rows -> conflict-free
  const int b  = blockIdx.z;
  const int c0 = blockIdx.y * 64;
  const int p0 = blockIdx.x * 64;
  const float mu = stats[2 * b], rs = stats[2 * b + 1];
  const int t = threadIdx.x;
#pragma unroll
  for (int i = 0; i < 16; ++i) {
    int lin = i * 256 + t;
    int cl = lin >> 6, pl = lin & 63;
    float v = (x[((size_t)(b * 256 + c0 + cl)) * HW + p0 + pl] - mu) * rs * gw[c0 + cl] + gb[c0 + cl];
    tile[cl * 66 + pl] = f2bf(v);
  }
  __syncthreads();
#pragma unroll
  for (int i = 0; i < 16; ++i) {
    int lin = i * 256 + t;
    int pl = lin >> 6, cl = lin & 63;
    out[((size_t)((b << 12) + p0 + pl)) * 256 + c0 + cl] = tile[cl * 66 + pl];
  }
}

// ---------- pack weights [N,K] f32 into per-lane WMMA fragment-ready bf16 ----------
__device__ __forceinline__ size_t packed_idx(int n, int k, int K) {
  int ntile = n >> 4, ks = k >> 5, kloc = k & 31;
  int g = (kloc >> 3) & 1;
  int j = (kloc & 7) | ((kloc >> 4) << 3);
  int lane = (n & 15) | (g << 4);
  return ((((size_t)ntile * (K >> 5)) + ks) * 32 + lane) * 16 + j;
}
__global__ void wpack(const float* __restrict__ in, unsigned short* __restrict__ out, int N, int K) {
  int idx = blockIdx.x * 256 + threadIdx.x;
  if (idx >= N * K) return;
  int n = idx / K, k = idx - n * K;
  out[packed_idx(n, k, K)] = f2bf(in[idx]);
}
// off1_w (O=256, I=512, 3, 3) f32 -> per-tap fragment-ready bf16 [tap][...]
__global__ void wpack3(const float* __restrict__ in, unsigned short* __restrict__ out) {
  int idx = blockIdx.x * 256 + threadIdx.x;        // 9*256*512
  if (idx >= 9 * 256 * 512) return;
  int tap = idx / (256 * 512);
  int rem = idx - tap * 256 * 512;
  int n = rem >> 9, k = rem & 511;
  float v = in[(size_t)(n * 512 + k) * 9 + tap];
  out[(size_t)tap * (256 * 512) + packed_idx(n, k, 512)] = f2bf(v);
}

// ---------- generic WMMA GEMM: out[P x N] = A[P x K] * Bw^T + bias ----------
// 64M x 128N macro-tile, 8 waves (wave = 16M x 64N). A staged in 64x256 K-slabs:
// one barrier pair per 8 K-steps -> 32 wmma per wave per slab. B fragment-ready from global.
// MODE 0: bf16 NHWC; MODE 2: exact GELU + bf16 NHWC; MODE 3: f32 NCHW out = resid + acc (N==256)
template <int MODE>
__global__ __launch_bounds__(256) void gemm_wmma(
    const unsigned short* __restrict__ A, const unsigned short* __restrict__ Bp,
    const float* __restrict__ bias, unsigned short* __restrict__ outb,
    float* __restrict__ outf, const float* __restrict__ resid, int K, int N) {
  __shared__ unsigned short ldsA[64 * SLAB_STR];

  const int t = threadIdx.x, lane = t & 31, wid = t >> 5;
  const int wm = wid & 3, wn = wid >> 2;          // 4 waves in M, 2 in N (64 cols each)
  const int r = t >> 2, ch = (t & 3) * 8;
  const size_t arow = (size_t)(blockIdx.x * 64 + r) * K;
  const int kst = K >> 5;                         // 32-wide K-steps
  const int nslab = K >> 8;                       // 256-wide K-slabs (K in {256,1024})
  const int nt0 = blockIdx.y * 8 + wn * 4;        // first 16-col tile for this wave
  const int arow_f = (lane & 15) + wm * 16, g = lane >> 4;

  v8f acc[4] = {{}, {}, {}, {}};

  for (int sl = 0; sl < nslab; ++sl) {
    __syncthreads();
#pragma unroll
    for (int j = 0; j < 8; ++j)
      *(uint4*)(ldsA + r * SLAB_STR + ch + j * 32) =
          *(const uint4*)(A + arow + sl * 256 + ch + j * 32);
    if (sl + 1 < nslab) __builtin_prefetch(A + arow + (sl + 1) * 256 + ch, 0, 1);
    __syncthreads();
#pragma unroll
    for (int ki = 0; ki < 8; ++ki) {
      const int ks = sl * 8 + ki;
      v16bf fa = frag_ld(ldsA, arow_f, g, ki);
#pragma unroll
      for (int s = 0; s < 4; ++s) {
        union { v8u u; v16bf b; } fb;
        fb.u = *(const v8u*)(Bp + ((((size_t)(nt0 + s)) * kst + ks) * 32 + lane) * 16);
        acc[s] = __builtin_amdgcn_wmma_f32_16x16x32_bf16(false, fa, false, fb.b, (short)0, acc[s], false, false);
      }
    }
  }

  // C/D layout: elem (vg,lane): m = vg + 8*(lane>>4), n = lane&15.
  const int mtop = blockIdx.x * 64 + wm * 16 + 8 * g;
#pragma unroll
  for (int s = 0; s < 4; ++s) {
    const int n = (lane & 15) + (nt0 + s) * 16;
    const float bs = bias[n];
#pragma unroll
    for (int vg = 0; vg < 8; ++vg) {
      const int mrow = mtop + vg;
      float v = acc[s][vg] + bs;
      if (MODE == 0) {
        outb[(size_t)mrow * N + n] = f2bf(v);
      } else if (MODE == 2) {
        v = 0.5f * v * (1.0f + erff(v * 0.70710678118654752f));
        outb[(size_t)mrow * N + n] = f2bf(v);
      } else { // MODE 3: NCHW residual store (channels == 256)
        const int b = mrow >> 12, p = mrow & 4095;
        const size_t o = ((size_t)(b * 256 + n)) * HW + p;
        outf[o] = resid[o] + v;
      }
    }
  }
}

// ---------- 3x3 conv (512 -> 256) as 9 shifted WMMA GEMM accumulations, ReLU ----------
// K=512 per tap = slab0 (xkvn channels) + slab1 (xqn channels); barriers once per slab.
__global__ __launch_bounds__(256) void conv3_wmma(
    const unsigned short* __restrict__ xkvn, const unsigned short* __restrict__ xqn,
    const unsigned short* __restrict__ w3p, const float* __restrict__ bias,
    unsigned short* __restrict__ outb) {
  __shared__ unsigned short ldsA[64 * SLAB_STR];

  const int t = threadIdx.x, lane = t & 31, wid = t >> 5;
  const int wm = wid & 3, wn = wid >> 2;
  const int r = t >> 2, ch = (t & 3) * 8;
  const int nt0 = blockIdx.y * 8 + wn * 4;
  const int arow_f = (lane & 15) + wm * 16, g = lane >> 4;

  const int mrow = blockIdx.x * 64 + r;
  const int b = mrow >> 12, p = mrow & 4095;
  const int y = p >> 6, x = p & 63;

  v8f acc[4] = {{}, {}, {}, {}};

  for (int tap = 0; tap < 9; ++tap) {
    const int ys = y + tap / 3 - 1, xs = x + tap % 3 - 1;
    const bool valid = ((unsigned)ys < 64u) && ((unsigned)xs < 64u);
    const size_t srcpix = ((size_t)((b << 12) + (ys << 6) + xs)) * 256;
    const unsigned short* wt = w3p + (size_t)tap * (256 * 512);

#pragma unroll
    for (int sl = 0; sl < 2; ++sl) {
      const unsigned short* src = (sl ? xqn : xkvn) + srcpix + ch;
      __syncthreads();
#pragma unroll
      for (int j = 0; j < 8; ++j) {
        uint4 va = {0u, 0u, 0u, 0u};
        if (valid) va = *(const uint4*)(src + j * 32);
        *(uint4*)(ldsA + r * SLAB_STR + ch + j * 32) = va;
      }
      __syncthreads();
#pragma unroll
      for (int ki = 0; ki < 8; ++ki) {
        const int ks = sl * 8 + ki;
        v16bf fa = frag_ld(ldsA, arow_f, g, ki);
#pragma unroll
        for (int s = 0; s < 4; ++s) {
          union { v8u u; v16bf b; } fb;
          fb.u = *(const v8u*)(wt + ((((size_t)(nt0 + s)) * 16 + ks) * 32 + lane) * 16);
          acc[s] = __builtin_amdgcn_wmma_f32_16x16x32_bf16(false, fa, false, fb.b, (short)0, acc[s], false, false);
        }
      }
    }
  }

  const int mtop = blockIdx.x * 64 + wm * 16 + 8 * g;
#pragma unroll
  for (int s = 0; s < 4; ++s) {
    const int n = (lane & 15) + (nt0 + s) * 16;
    const float bs = bias[n];
#pragma unroll
    for (int vg = 0; vg < 8; ++vg) {
      float v = acc[s][vg] + bs;
      v = v > 0.f ? v : 0.f;   // ReLU
      outb[(size_t)(mtop + vg) * 256 + n] = f2bf(v);
    }
  }
}

// ---------- tiny 256->2 conv producing sample coords; one wave32 per pixel ----------
__global__ void off2_kernel(const unsigned short* __restrict__ h, const float* __restrict__ w2,
                            const float* __restrict__ b2, float* __restrict__ offxy) {
  __shared__ float sw[512];
  for (int i = threadIdx.x; i < 512; i += 256) sw[i] = w2[i];
  __syncthreads();
  const int lane = threadIdx.x & 31, wid = threadIdx.x >> 5;
  const int pp = blockIdx.x * 8 + wid;            // NPIX/8 blocks
  const unsigned short* hp = h + (size_t)pp * 256;
  float a0 = 0.f, a1 = 0.f;
#pragma unroll
  for (int i = 0; i < 8; ++i) {
    int c = lane + i * 32;
    float v = bf2f(hp[c]);
    a0 += v * sw[c];
    a1 += v * sw[256 + c];
  }
#pragma unroll
  for (int m = 16; m > 0; m >>= 1) {
    a0 += __shfl_xor(a0, m, 32);
    a1 += __shfl_xor(a1, m, 32);
  }
  if (lane == 0) {
    offxy[2 * pp]     = a0 + b2[0];   // ix (grid math cancels: ((off/(W-1)*2-1)+1)*.5*(W-1) == off)
    offxy[2 * pp + 1] = a1 + b2[1];   // iy
  }
}

// ---------- bilinear grid-sample with zero padding, from NHWC bf16 ----------
__global__ void gsample(const unsigned short* __restrict__ src, const float* __restrict__ offxy,
                        unsigned short* __restrict__ out) {
  const int bp = blockIdx.x;        // pixel
  const int c = threadIdx.x;        // channel
  const int b = bp >> 12;
  const float ix = offxy[2 * bp], iy = offxy[2 * bp + 1];
  const float x0f = floorf(ix), y0f = floorf(iy);
  const float wx1 = ix - x0f, wx0 = 1.f - wx1;
  const float wy1 = iy - y0f, wy0 = 1.f - wy1;
  const int x0 = (int)x0f, y0 = (int)y0f;
  float acc = 0.f;
#pragma unroll
  for (int k = 0; k < 4; ++k) {
    const int yy = y0 + (k >> 1), xx = x0 + (k & 1);
    const float w = ((k >> 1) ? wy1 : wy0) * ((k & 1) ? wx1 : wx0);
    if ((unsigned)yy < 64u && (unsigned)xx < 64u)
      acc += w * bf2f(src[((size_t)((b << 12) + (yy << 6) + xx)) * 256 + c]);
  }
  out[(size_t)bp * 256 + c] = f2bf(acc);
}

// ---------- gfx1250 feature probe: cluster id/barrier + NULL-descriptor TDM ----------
__global__ void cdna5_probe(int* __restrict__ sink) {
  __shared__ unsigned char lds[256];
  int cid = 0;
#if __has_builtin(__builtin_amdgcn_cluster_id_x)
  cid = __builtin_amdgcn_cluster_id_x();
#endif
#if __has_builtin(__builtin_amdgcn_s_cluster_barrier)
  __builtin_amdgcn_s_cluster_barrier();
#endif
#if __has_builtin(__builtin_amdgcn_tensor_load_to_lds)
  u32x4 g0 = {0u, 0u, 0u, 0u};                 // count=0 -> NULL descriptor: architectural no-op
  i32x8 g1 = {0, 0, 0, 0, 0, 0, 0, 0};
  i32x4 g2 = {0, 0, 0, 0};
  i32x4 g3 = {0, 0, 0, 0};
 #if defined(__clang_major__) && (__clang_major__ >= 23)
  i32x8 g4 = {0, 0, 0, 0, 0, 0, 0, 0};
  __builtin_amdgcn_tensor_load_to_lds(g0, g1, g2, g3, g4, 0);
 #else
  __builtin_amdgcn_tensor_load_to_lds(g0, g1, g2, g3, 0);
 #endif
#endif
#if __has_builtin(__builtin_amdgcn_s_wait_tensorcnt)
  __builtin_amdgcn_s_wait_tensorcnt(0);
#endif
  lds[threadIdx.x] = (unsigned char)cid;
  __syncthreads();
  if (threadIdx.x == 0) sink[0] = cid + (int)lds[31];
}

// =====================================================================
extern "C" void kernel_launch(void* const* d_in, const int* in_sizes, int n_in,
                              void* d_out, int out_size, void* d_ws, size_t ws_size,
                              hipStream_t stream) {
  const float* x_q    = (const float*)d_in[0];
  const float* x_kv   = (const float*)d_in[1];
  const float* n1_w   = (const float*)d_in[2];
  const float* n1_b   = (const float*)d_in[3];
  // q_w/q_b (d_in[4..5]) mathematically dead: softmax over a size-1 axis == 1
  const float* kv_w   = (const float*)d_in[6];
  const float* kv_b   = (const float*)d_in[7];
  const float* off1_w = (const float*)d_in[8];
  const float* off1_b = (const float*)d_in[9];
  const float* off2_w = (const float*)d_in[10];
  const float* off2_b = (const float*)d_in[11];
  const float* proj_w = (const float*)d_in[12];
  const float* proj_b = (const float*)d_in[13];
  const float* n2_w   = (const float*)d_in[14];
  const float* n2_b   = (const float*)d_in[15];
  const float* fc1_w  = (const float*)d_in[16];
  const float* fc1_b  = (const float*)d_in[17];
  const float* fc2_w  = (const float*)d_in[18];
  const float* fc2_b  = (const float*)d_in[19];
  float* out = (float*)d_out;

  // workspace carve-up (256B aligned)
  char* w = (char*)d_ws;
  auto alloc = [&](size_t bytes) { char* p = w; w += (bytes + 255) & ~(size_t)255; return p; };
  float*          sQ    = (float*)         alloc(256);
  float*          sKV   = (float*)         alloc(256);
  float*          sX1   = (float*)         alloc(256);
  int*            probe = (int*)           alloc(256);
  unsigned short* xqn   = (unsigned short*)alloc((size_t)NPIX * 256 * 2);
  unsigned short* xkvn  = (unsigned short*)alloc((size_t)NPIX * 256 * 2);
  unsigned short* w3p   = (unsigned short*)alloc((size_t)9 * 256 * 512 * 2);
  unsigned short* kvvw  = (unsigned short*)alloc((size_t)256 * 256 * 2);
  unsigned short* projw = (unsigned short*)alloc((size_t)256 * 256 * 2);
  unsigned short* fc1wb = (unsigned short*)alloc((size_t)1024 * 256 * 2);
  unsigned short* fc2wb = (unsigned short*)alloc((size_t)256 * 1024 * 2);
  unsigned short* hbuf  = (unsigned short*)alloc((size_t)NPIX * 256 * 2);
  float*          offxy = (float*)         alloc((size_t)NPIX * 2 * 4);
  unsigned short* samp  = (unsigned short*)alloc((size_t)NPIX * 256 * 2);
  unsigned short* vbuf  = (unsigned short*)alloc((size_t)NPIX * 256 * 2);
  float*          x1    = (float*)         alloc((size_t)NPIX * 256 * 4);
  unsigned short* x1n   = (unsigned short*)alloc((size_t)NPIX * 256 * 2);
  unsigned short* gbuf  = (unsigned short*)alloc((size_t)NPIX * 1024 * 2);

  cdna5_probe<<<1, 32, 0, stream>>>(probe);

  // 1) per-batch stats + bf16 NHWC normalization of x_q, x_kv
  stats_kernel<<<8, 256, 0, stream>>>(x_q,  sQ);
  stats_kernel<<<8, 256, 0, stream>>>(x_kv, sKV);
  norm_pack<<<dim3(64, 4, 8), 256, 0, stream>>>(x_q,  sQ,  n1_w, n1_b, xqn);
  norm_pack<<<dim3(64, 4, 8), 256, 0, stream>>>(x_kv, sKV, n1_w, n1_b, xkvn);

  // 2) fragment-ready weight packing (v-half of kv_w = rows 256..511)
  wpack<<<256,  256, 0, stream>>>(proj_w,           projw, 256, 256);
  wpack<<<256,  256, 0, stream>>>(kv_w + 256 * 256, kvvw,  256, 256);
  wpack<<<1024, 256, 0, stream>>>(fc1_w,            fc1wb, 1024, 256);
  wpack<<<1024, 256, 0, stream>>>(fc2_w,            fc2wb, 256, 1024);
  wpack3<<<4608, 256, 0, stream>>>(off1_w, w3p);

  // 3) offset branch: 3x3 conv (concat(xkvn,xqn) -> 256) + ReLU, then 256->2
  conv3_wmma<<<dim3(512, 2), 256, 0, stream>>>(xkvn, xqn, w3p, off1_b, hbuf);
  off2_kernel<<<4096, 256, 0, stream>>>(hbuf, off2_w, off2_b, offxy);

  // 4) bilinear sample of inner_kv (= xq_n) at the predicted coords
  gsample<<<32768, 256, 0, stream>>>(xqn, offxy, samp);

  // 5) v = kv-conv(sampled)[256:512]  (attn == 1 exactly), then proj + residual -> x1 (NCHW f32)
  gemm_wmma<0><<<dim3(512, 2), 256, 0, stream>>>(samp, kvvw, kv_b + 256, vbuf, nullptr, nullptr, 256, 256);
  gemm_wmma<3><<<dim3(512, 2), 256, 0, stream>>>(vbuf, projw, proj_b, nullptr, x1, x_q, 256, 256);

  // 6) norm2, fc1 + exact GELU, fc2 + residual -> out (NCHW f32)
  stats_kernel<<<8, 256, 0, stream>>>(x1, sX1);
  norm_pack<<<dim3(64, 4, 8), 256, 0, stream>>>(x1, sX1, n2_w, n2_b, x1n);
  gemm_wmma<2><<<dim3(512, 8), 256, 0, stream>>>(x1n, fc1wb, fc1_b, gbuf, nullptr, nullptr, 256, 1024);
  gemm_wmma<3><<<dim3(512, 2), 256, 0, stream>>>(gbuf, fc2wb, fc2_b, nullptr, out, x1, 1024, 256);
}
